// GCNEncoder_84731114816416
// MI455X (gfx1250) — matrix operand order repaired and checked by reading, not compiled, not run on hardware
//
#include <hip/hip_runtime.h>
#include <hip/hip_bf16.h>

// ---- CDNA5 WMMA vector types -------------------------------------------------
typedef __attribute__((ext_vector_type(16))) _Float16 v16h;
typedef __attribute__((ext_vector_type(8)))  float    v8f;

#define DHF 128   // feature width (DIN == DH == 128)

// ------------------------------------------------------- zero buffer (float4)
__global__ void k_zero4(float4* __restrict__ p, int n4) {
    int i = blockIdx.x * blockDim.x + threadIdx.x;
    if (i < n4) p[i] = make_float4(0.f, 0.f, 0.f, 0.f);
}

// ------------------------------------------------------------- degree (atomic)
__global__ void k_degree(const int* __restrict__ dst, float* __restrict__ deg, int E) {
    int e = blockIdx.x * blockDim.x + threadIdx.x;
    if (e < E) atomicAdd(&deg[dst[e]], 1.0f);
}

// -------------------------------------------------- dinv = rsqrt(deg+1) inplace
__global__ void k_rsqrt1(float* __restrict__ deg, int N) {
    int n = blockIdx.x * blockDim.x + threadIdx.x;
    if (n < N) deg[n] = rsqrtf(deg[n] + 1.0f);
}

// ----------------------------------------------------------- fp32 -> f16 weights
__global__ void k_cvt_f16(const float* __restrict__ w, _Float16* __restrict__ wh, int n) {
    int i = blockIdx.x * blockDim.x + threadIdx.x;
    if (i < n) wh[i] = (_Float16)w[i];
}

// -----------------------------------------------------------------------------
// H = X @ W^T   (X: [N,128] fp32, W given as f16 [128out,128in] row-major)
// One block = 256 threads = 8 waves; block covers 16 rows x 128 cols.
// Wave w computes the 16x16 tile at cols [16w, 16w+16) with 4x
// V_WMMA_F32_16X16X32_F16 over K=128.
// Fragment packing per ISA 7.12.2 (wave32):
//   A (16-bit 16x32): lane%16 = M row; VGPRs 0-3 hold K = hsel*8 + 0..7,
//                     VGPRs 4-7 hold K = 16 + hsel*8 + 0..7  (contiguous 16B runs)
//   B (32x16):        lane%16 = N column; same K order as A (VGPRs 0-3: hsel*8 + 0..7, 4-7: 16 + hsel*8 + 0..7)
//   D (f32 16x16):    lane%16 = N, VGPR r -> M = r + 8*hsel
// -----------------------------------------------------------------------------
__global__ __launch_bounds__(256) void k_gemm_wmma(
        const float* __restrict__ X, const _Float16* __restrict__ Wh,
        float* __restrict__ H, int N) {
    __shared__ _Float16 Xs[16][136];           // +16B pad per row vs LDS banks

    const int tid     = threadIdx.x;
    const int rowbase = blockIdx.x * 16;

    // Stage the 16x128 A tile into LDS as f16 (clamped row -> no EXEC holes).
    for (int i = tid; i < 16 * DHF; i += 256) {
        int r = i >> 7, c = i & 127;
        int gr = rowbase + r; if (gr >= N) gr = N - 1;
        Xs[r][c] = (_Float16)X[(size_t)gr * DHF + c];
    }
    __syncthreads();

    const int wave    = tid >> 5;
    const int lane    = tid & 31;
    const int hsel    = lane >> 4;     // lane half (0: lanes 0-15, 1: 16-31)
    const int rc      = lane & 15;     // row (A) / column (B,D) within tile
    const int colbase = wave * 16;

    v8f acc = {};
#pragma unroll
    for (int k0 = 0; k0 < DHF; k0 += 32) {
        v16h a, b;
#pragma unroll
        for (int p = 0; p < 8; ++p) {
            const int ka = ((p < 4) ? 0 : 16) + hsel * 8 + 2 * (p & 3);
            a[2 * p]     = Xs[rc][k0 + ka];
            a[2 * p + 1] = Xs[rc][k0 + ka + 1];
            const int kb = ((p < 4) ? 0 : 16) + hsel * 8 + 2 * (p & 3);
            const size_t wrow = (size_t)(colbase + rc) * DHF + (size_t)k0 + kb;
            b[2 * p]     = Wh[wrow];
            b[2 * p + 1] = Wh[wrow + 1];
        }
        // emits v_wmma_f32_16x16x32_f16
        acc = __builtin_amdgcn_wmma_f32_16x16x32_f16(
                  /*neg_a=*/false, a, /*neg_b=*/false, b,
                  /*c_mod=*/(short)0, acc, /*reuse_a=*/false, /*reuse_b=*/false);
    }

    // Block-uniform bounds check: hot path has zero EXEC manipulation.
    float* Hp = H + (size_t)rowbase * DHF + colbase + rc;
    if (rowbase + 16 <= N) {
#pragma unroll
        for (int r = 0; r < 8; ++r)
            Hp[(size_t)(r + 8 * hsel) * DHF] = acc[r];
    } else {
#pragma unroll
        for (int r = 0; r < 8; ++r) {
            const int m = r + 8 * hsel;
            if (rowbase + m < N) Hp[(size_t)m * DHF] = acc[r];
        }
    }
}

// -----------------------------------------------------------------------------
// Edge scatter: AGG[dst] += H[src] * (dinv[src]*dinv[dst]).
// 32 lanes per edge; each lane moves a float4 (L2-resident gather, fp32 atomics).
// global_prefetch warms the destination line before the atomic burst.
// -----------------------------------------------------------------------------
__global__ __launch_bounds__(256) void k_scatter(
        const float* __restrict__ H,
        const int* __restrict__ src, const int* __restrict__ dst,
        const float* __restrict__ dinv, float* __restrict__ AGG, int E) {
    const int e = blockIdx.x * 8 + (threadIdx.x >> 5);
    if (e >= E) return;
    const int lane = threadIdx.x & 31;
    const int s = src[e], d = dst[e];

    float* o = AGG + (size_t)d * DHF + lane * 4;
    __builtin_prefetch(o, 1, 3);                       // global_prefetch_b8

    const float norm = dinv[s] * dinv[d];
    const float4 v = ((const float4*)(H + (size_t)s * DHF))[lane];
    atomicAdd(o + 0, v.x * norm);
    atomicAdd(o + 1, v.y * norm);
    atomicAdd(o + 2, v.z * norm);
    atomicAdd(o + 3, v.w * norm);
}

// --------------------------- layer-0 epilogue: +selfloop +bias, dropout, ReLU
__global__ void k_combine_relu4(
        const float4* __restrict__ AGG, const float4* __restrict__ H,
        const float* __restrict__ dinv, const float4* __restrict__ bias,
        const float4* __restrict__ mask, float4* __restrict__ hout, int total4) {
    int i = blockIdx.x * blockDim.x + threadIdx.x;
    if (i >= total4) return;
    const int n = i >> 5, f4 = i & 31;                 // 32 float4 per row
    const float di = dinv[n], d2 = di * di;
    const float4 a = AGG[i], h = H[i], b = bias[f4], m = mask[i];
    float4 v;
    v.x = fmaxf((a.x + h.x * d2 + b.x) * m.x, 0.f);
    v.y = fmaxf((a.y + h.y * d2 + b.y) * m.y, 0.f);
    v.z = fmaxf((a.z + h.z * d2 + b.z) * m.z, 0.f);
    v.w = fmaxf((a.w + h.w * d2 + b.w) * m.w, 0.f);
    hout[i] = v;
}

// -------------------------------- layer-1/2 epilogue: +selfloop +bias -> out
__global__ void k_combine_out4(
        const float4* __restrict__ AGG, const float4* __restrict__ H,
        const float* __restrict__ dinv, const float4* __restrict__ bias,
        float4* __restrict__ out, int total4) {
    int i = blockIdx.x * blockDim.x + threadIdx.x;
    if (i >= total4) return;
    const int n = i >> 5, f4 = i & 31;
    const float di = dinv[n], d2 = di * di;
    const float4 a = AGG[i], h = H[i], b = bias[f4];
    float4 v;
    v.x = a.x + h.x * d2 + b.x;
    v.y = a.y + h.y * d2 + b.y;
    v.z = a.z + h.z * d2 + b.z;
    v.w = a.w + h.w * d2 + b.w;
    out[i] = v;
}

// -----------------------------------------------------------------------------
extern "C" void kernel_launch(void* const* d_in, const int* in_sizes, int n_in,
                              void* d_out, int out_size, void* d_ws, size_t ws_size,
                              hipStream_t stream) {
    const float* x    = (const float*)d_in[0];
    const int*   eidx = (const int*)  d_in[1];
    const float* mask = (const float*)d_in[2];
    const float* W0   = (const float*)d_in[3];
    const float* b0   = (const float*)d_in[4];
    const float* W1   = (const float*)d_in[5];
    const float* b1   = (const float*)d_in[6];
    const float* W2   = (const float*)d_in[7];
    const float* b2   = (const float*)d_in[8];

    const int N = in_sizes[0] / DHF;
    const int E = in_sizes[1] / 2;
    const int* src = eidx;
    const int* dst = eidx + E;
    const size_t NF = (size_t)N * DHF;
    const int NF4 = (int)(NF / 4);

    // ---- workspace carve-up (256B aligned slots) ----
    char* ws = (char*)d_ws;
    size_t off = 0;
    auto take = [&](size_t bytes) -> void* {
        void* p = ws + off;
        off += (bytes + 255) & ~(size_t)255;
        return p;
    };
    float*    dinv = (float*)   take((size_t)N * sizeof(float));    // deg -> dinv in place
    _Float16* Wh0  = (_Float16*)take((size_t)DHF * DHF * 2);
    _Float16* Wh1  = (_Float16*)take((size_t)DHF * DHF * 2);
    _Float16* Wh2  = (_Float16*)take((size_t)DHF * DHF * 2);
    float*    H    = (float*)   take(NF * sizeof(float));           // pre-agg linear output
    float*    hact = (float*)   take(NF * sizeof(float));           // post dropout+relu
    float*    AGG  = (float*)   take(NF * sizeof(float));           // neighbor sum

    float* out0 = (float*)d_out;        // x_
    float* out1 = out0 + NF;            // x2

    const int T = 256;
    const int gN    = (N + T - 1) / T;
    const int gN4   = ((N / 4) + T - 1) / T;          // N divisible by 4 here
    const int gE    = (E + T - 1) / T;
    const int gNF4  = (NF4 + T - 1) / T;
    const int gW    = (DHF * DHF + T - 1) / T;
    const int gGemm = (N + 15) / 16;
    const int gScat = (E + 7) / 8;

    // degrees -> dinv  (shared by all 3 layers)
    k_zero4 <<<gN4, T, 0, stream>>>((float4*)dinv, N / 4);
    k_degree<<<gE,  T, 0, stream>>>(dst, dinv, E);
    k_rsqrt1<<<gN,  T, 0, stream>>>(dinv, N);

    // weights to f16 once per launch
    k_cvt_f16<<<gW, T, 0, stream>>>(W0, Wh0, DHF * DHF);
    k_cvt_f16<<<gW, T, 0, stream>>>(W1, Wh1, DHF * DHF);
    k_cvt_f16<<<gW, T, 0, stream>>>(W2, Wh2, DHF * DHF);

    // ---- layer 0: h = relu(drop_mask * gcn(x, W0, b0)) ----
    k_gemm_wmma    <<<gGemm, T, 0, stream>>>(x, Wh0, H, N);
    k_zero4        <<<gNF4,  T, 0, stream>>>((float4*)AGG, NF4);
    k_scatter      <<<gScat, T, 0, stream>>>(H, src, dst, dinv, AGG, E);
    k_combine_relu4<<<gNF4,  T, 0, stream>>>((const float4*)AGG, (const float4*)H, dinv,
                                             (const float4*)b0, (const float4*)mask,
                                             (float4*)hact, NF4);

    // ---- layer 1: x_ = gcn(h, W1, b1) ----
    k_gemm_wmma   <<<gGemm, T, 0, stream>>>(hact, Wh1, H, N);
    k_zero4       <<<gNF4,  T, 0, stream>>>((float4*)AGG, NF4);
    k_scatter     <<<gScat, T, 0, stream>>>(H, src, dst, dinv, AGG, E);
    k_combine_out4<<<gNF4,  T, 0, stream>>>((const float4*)AGG, (const float4*)H, dinv,
                                            (const float4*)b1, (float4*)out0, NF4);

    // ---- layer 2: x2 = gcn(h, W2, b2) ----
    k_gemm_wmma   <<<gGemm, T, 0, stream>>>(hact, Wh2, H, N);
    k_zero4       <<<gNF4,  T, 0, stream>>>((float4*)AGG, NF4);
    k_scatter     <<<gScat, T, 0, stream>>>(H, src, dst, dinv, AGG, E);
    k_combine_out4<<<gNF4,  T, 0, stream>>>((const float4*)AGG, (const float4*)H, dinv,
                                            (const float4*)b2, (float4*)out1, NF4);
}
